// MultiHeadAttention_24051816858328
// MI455X (gfx1250) — matrix operand order repaired and checked by reading, not compiled, not run on hardware
//
#include <hip/hip_runtime.h>

// ---------------------------------------------------------------------------
// MHA forward for MI455X (gfx1250, wave32, WMMA bf16)
// B=4, LQ=LK=2048, D_MODEL=1024, H=16, HD=64
// ---------------------------------------------------------------------------

#define DEVINL __device__ __forceinline__

typedef __attribute__((ext_vector_type(16))) __bf16 v16bf;
typedef __attribute__((ext_vector_type(8)))  float  v8f;

union FragU {
  v16bf v;
  unsigned short u[16];
  uint4 q[2];
};

static constexpr int BATCH   = 4;
static constexpr int SEQ     = 2048;
static constexpr int DMODEL  = 1024;
static constexpr int NHEADS  = 16;
static constexpr int HDIM    = 64;
static constexpr int ROWS    = BATCH * SEQ;                       // 8192
static constexpr long HBUF   = (long)BATCH * NHEADS * SEQ * HDIM; // 8388608
static constexpr long WBUF   = (long)DMODEL * DMODEL;             // 1048576

DEVINL unsigned short f2bf(float x) {
  unsigned int b = __builtin_bit_cast(unsigned int, x);
  b += 0x7FFFu + ((b >> 16) & 1u);            // round to nearest even
  return (unsigned short)(b >> 16);
}

DEVINL v8f wmma_bf16(v16bf a, v16bf b, v8f c) {
  return __builtin_amdgcn_wmma_f32_16x16x32_bf16(
      /*neg_a=*/false, a, /*neg_b=*/false, b,
      /*c_mod=*/(short)0, c, /*reuse_a=*/false, /*reuse_b=*/false);
}

// A fragment (16x32, MxK): lane 0-15 row M=lane, K runs [kk,kk+7],[16+kk,16+kk+7], kk = lane<16?0:8
DEVINL v16bf afrag_bf16(const unsigned short* A, int ld, int row0, int k0) {
  int lane = threadIdx.x & 31;
  int r  = row0 + (lane & 15);
  int kk = (lane & 16) ? 8 : 0;
  const unsigned short* p = A + (long)r * ld + k0 + kk;
  FragU f;
  f.q[0] = *(const uint4*)(p);
  f.q[1] = *(const uint4*)(p + 16);
  return f.v;
}

DEVINL v16bf afrag_f32(const float* A, int ld, int row0, int k0) {
  int lane = threadIdx.x & 31;
  int r  = row0 + (lane & 15);
  int kk = (lane & 16) ? 8 : 0;
  const float* p = A + (long)r * ld + k0 + kk;
  FragU f;
#pragma unroll
  for (int j = 0; j < 8; ++j) f.u[j]     = f2bf(p[j]);
#pragma unroll
  for (int j = 0; j < 8; ++j) f.u[8 + j] = f2bf(p[16 + j]);
  return f.v;
}

// B fragment (32x16, KxN) from Bt laid out [n][k] (row n contiguous in k):
// lane 0-15 column N=lane, K runs [kk,kk+7],[16+kk,16+kk+7], kk = lane<16?0:8
DEVINL v16bf bfrag_bf16(const unsigned short* Bt, int ld, int col0, int k0) {
  int lane = threadIdx.x & 31;
  int n  = col0 + (lane & 15);
  int kk = (lane & 16) ? 8 : 0;
  const unsigned short* p = Bt + (long)n * ld + k0 + kk;
  FragU f;
  f.q[0] = *(const uint4*)(p);
  f.q[1] = *(const uint4*)(p + 16);
  return f.v;
}

// ---------------------------------------------------------------------------
// Kernel 0: one-shot f32 -> bf16 weight conversion (each thread: 8 elements)
// ---------------------------------------------------------------------------
__global__ __launch_bounds__(256) void bf16conv_kernel(
    const float* __restrict__ src, unsigned short* __restrict__ dst, int n8) {
  int idx = blockIdx.x * 256 + threadIdx.x;
  if (idx >= n8) return;
  const float4* s = (const float4*)src + (long)idx * 2;
  float4 a = s[0], b = s[1];
  FragU f;
  f.u[0] = f2bf(a.x); f.u[1] = f2bf(a.y); f.u[2] = f2bf(a.z); f.u[3] = f2bf(a.w);
  f.u[4] = f2bf(b.x); f.u[5] = f2bf(b.y); f.u[6] = f2bf(b.z); f.u[7] = f2bf(b.w);
  ((uint4*)dst)[idx] = f.q[0];
}

// ---------------------------------------------------------------------------
// Kernel 1: projection  Y = X @ W^T + b  (+RoPE, +scale), bf16 head output.
// W pre-converted to bf16 (B-frags are pure b128 loads).
// MODE 0: Q -> qh[b,h,l,d] (rope, *0.125)   MODE 1: K -> kh[b,h,l,d] (rope)
// MODE 2: V -> vt[b,h,d,l] (transposed)
// block = 256 thr = 8 waves; wave = 16 rows x 64 cols; grid = (64, 16)
// ---------------------------------------------------------------------------
template <int MODE>
__global__ __launch_bounds__(256) void proj_kernel(
    const float* __restrict__ X, const unsigned short* __restrict__ Wb,
    const float* __restrict__ bias,
    const float* __restrict__ cosT, const float* __restrict__ sinT,
    unsigned short* __restrict__ out) {
  int wave = threadIdx.x >> 5;
  int lane = threadIdx.x & 31;
  int row0 = blockIdx.x * 128 + wave * 16;
  int col0 = blockIdx.y * 64;           // head-aligned strip (head = blockIdx.y)
  int h    = blockIdx.y;

  v8f acc[4] = {};
  for (int k0 = 0; k0 < DMODEL; k0 += 32) {
    v16bf a = afrag_f32(X, DMODEL, row0, k0);
#pragma unroll
    for (int t = 0; t < 4; ++t) {
      v16bf b = bfrag_bf16(Wb, DMODEL, col0 + t * 16, k0);
      acc[t] = wmma_bf16(a, b, acc[t]);
    }
  }

  int nlo   = lane & 15;
  int rbase = row0 + ((lane & 16) ? 8 : 0);

  float vals[4][8];
#pragma unroll
  for (int t = 0; t < 4; ++t)
#pragma unroll
    for (int i = 0; i < 8; ++i)
      vals[t][i] = acc[t][i] + bias[col0 + t * 16 + nlo];

  if (MODE < 2) {  // RoPE: rot[d] = (d<32) ? -y[d+32] : y[d-32]
    float rot[4][8];
#pragma unroll
    for (int t = 0; t < 4; ++t)
#pragma unroll
      for (int i = 0; i < 8; ++i)
        rot[t][i] = (t < 2) ? -vals[t + 2][i] : vals[t - 2][i];
#pragma unroll
    for (int t = 0; t < 4; ++t) {
      int d = t * 16 + nlo;
#pragma unroll
      for (int i = 0; i < 8; ++i) {
        int row = rbase + i;
        int b = row >> 11, l = row & (SEQ - 1);
        long ci = ((long)b * SEQ + l) * HDIM + d;
        float c = cosT[ci], s = sinT[ci];
        float v = vals[t][i] * c + rot[t][i] * s;
        if (MODE == 0) v *= 0.125f;    // fold 1/sqrt(HDIM) into Q
        vals[t][i] = v;
      }
    }
  }

#pragma unroll
  for (int t = 0; t < 4; ++t) {
    int d = t * 16 + nlo;
#pragma unroll
    for (int i = 0; i < 8; ++i) {
      int row = rbase + i;
      int b = row >> 11, l = row & (SEQ - 1);
      long idx;
      if (MODE == 2)
        idx = (((long)b * NHEADS + h) * HDIM + d) * SEQ + l;     // vt[b,h,d,l]
      else
        idx = (((long)b * NHEADS + h) * SEQ + l) * HDIM + d;     // qh/kh[b,h,l,d]
      out[idx] = f2bf(vals[t][i]);
    }
  }
}

// ---------------------------------------------------------------------------
// Kernel 2: flash attention. Wave = 16 q-rows x full head (64), keys in
// chunks of 32. Scores via 4 WMMA; raw f32 scores are transposed through a
// wave-private padded LDS tile so softmax runs in A-fragment layout: each
// lane owns half of row (lane&15), so row max/sum are in-lane reductions
// plus ONE shfl_xor(16). exp() results are packed into the bf16 P A-frag in
// registers (no second LDS bounce); only the per-row rescale factor is
// broadcast back to C-layout (8 bpermutes). P@V via 4 WMMA vs transposed V.
// grid = (16, 64), block = 256.
// ---------------------------------------------------------------------------
__global__ __launch_bounds__(256) void attn_kernel(
    const unsigned short* __restrict__ qh, const unsigned short* __restrict__ kh,
    const unsigned short* __restrict__ vt, const unsigned char* __restrict__ mask,
    unsigned short* __restrict__ oh) {
  __shared__ __align__(16) float sst[8][16][36];   // stride 36: 16B-aligned, bank-spread

  int wave = threadIdx.x >> 5;
  int lane = threadIdx.x & 31;
  int nlo  = lane & 15;
  int hi   = (lane & 16) ? 1 : 0;
  int bh   = blockIdx.y;
  int b = bh >> 4, h = bh & 15;
  int q0 = blockIdx.x * 128 + wave * 16;

  const unsigned short* Q = qh + (long)bh * SEQ * HDIM;
  const unsigned short* K = kh + (long)bh * SEQ * HDIM;
  const unsigned short* V = vt + (long)bh * HDIM * SEQ;

  v16bf aq0 = afrag_bf16(Q, HDIM, q0, 0);
  v16bf aq1 = afrag_bf16(Q, HDIM, q0, 32);

  v8f o[4] = {};
  float m = -3.0e38f, l = 0.0f;          // row state, row = lane&15 (dup in halves)

  for (int kt = 0; kt < SEQ; kt += 32) {
    v8f s0 = {}, s1 = {};
    s0 = wmma_bf16(aq0, bfrag_bf16(K, HDIM, kt, 0),       s0);
    s0 = wmma_bf16(aq1, bfrag_bf16(K, HDIM, kt, 32),      s0);
    s1 = wmma_bf16(aq0, bfrag_bf16(K, HDIM, kt + 16, 0),  s1);
    s1 = wmma_bf16(aq1, bfrag_bf16(K, HDIM, kt + 16, 32), s1);

    bool mk0 = mask[b * SEQ + kt + nlo] != 0;
    bool mk1 = mask[b * SEQ + kt + 16 + nlo] != 0;

    // masked raw scores, C-layout -> LDS [16 rows][32 keys]
    int rr = hi ? 8 : 0;
#pragma unroll
    for (int i = 0; i < 8; ++i) {
      sst[wave][rr + i][nlo]      = mk0 ? -3.0e38f : s0[i];
      sst[wave][rr + i][16 + nlo] = mk1 ? -3.0e38f : s1[i];
    }
    asm volatile("s_wait_dscnt 0x0" ::: "memory");  // wave-internal LDS RAW

    // transpose read straight into A-frag k-order: lane owns row nlo,
    // k in {kk..kk+7} u {16+kk..16+kk+7}, kk = 8*hi
    const float* rp = &sst[wave][nlo][0];
    int kk = hi ? 8 : 0;
    float4 c0 = *(const float4*)(rp + kk);
    float4 c1 = *(const float4*)(rp + kk + 4);
    float4 c2 = *(const float4*)(rp + 16 + kk);
    float4 c3 = *(const float4*)(rp + 16 + kk + 4);
    float sv[16] = {c0.x, c0.y, c0.z, c0.w, c1.x, c1.y, c1.z, c1.w,
                    c2.x, c2.y, c2.z, c2.w, c3.x, c3.y, c3.z, c3.w};

    // row max: in-lane tree + one xor-16 (partner lane holds other k half)
    float lm = sv[0];
#pragma unroll
    for (int j = 1; j < 16; ++j) lm = fmaxf(lm, sv[j]);
    lm = fmaxf(lm, __shfl_xor(lm, 16, 32));
    float nm = fmaxf(m, lm);
    float sc = __expf(m - nm);
    m = nm;

    // exp + row sum + pack bf16 P A-frag in-lane
    FragU ap;
    float ls = 0.0f;
#pragma unroll
    for (int j = 0; j < 16; ++j) {
      float e = __expf(sv[j] - nm);
      ls += e;
      ap.u[j] = f2bf(e);
    }
    ls += __shfl_xor(ls, 16, 32);
    l = l * sc + ls;

    // broadcast per-row rescale factor into C-layout (row = i + 8*hi)
    float osc[8];
#pragma unroll
    for (int i = 0; i < 8; ++i) osc[i] = __shfl(sc, i + (hi ? 8 : 0), 32);
#pragma unroll
    for (int t = 0; t < 4; ++t)
#pragma unroll
      for (int i = 0; i < 8; ++i) o[t][i] *= osc[i];

#pragma unroll
    for (int t = 0; t < 4; ++t)
      o[t] = wmma_bf16(ap.v, bfrag_bf16(V, SEQ, t * 16, kt), o[t]);
  }

  // epilogue: broadcast per-row 1/l into C-layout and store
  float linv[8];
#pragma unroll
  for (int i = 0; i < 8; ++i) linv[i] = 1.0f / __shfl(l, i + (hi ? 8 : 0), 32);
#pragma unroll
  for (int i = 0; i < 8; ++i) {
    int q = q0 + (hi ? 8 : 0) + i;
    long orow = (long)b * SEQ + q;
#pragma unroll
    for (int t = 0; t < 4; ++t)
      oh[orow * DMODEL + h * HDIM + t * 16 + nlo] = f2bf(o[t][i] * linv[i]);
  }
}

// ---------------------------------------------------------------------------
// Kernel 3: output projection  Y = oh @ Wo^T + bo  (f32 out, Wo pre-converted)
// ---------------------------------------------------------------------------
__global__ __launch_bounds__(256) void oproj_kernel(
    const unsigned short* __restrict__ A, const unsigned short* __restrict__ Wob,
    const float* __restrict__ bo, float* __restrict__ Y) {
  int wave = threadIdx.x >> 5;
  int lane = threadIdx.x & 31;
  int row0 = blockIdx.x * 128 + wave * 16;
  int col0 = blockIdx.y * 64;

  v8f acc[4] = {};
  for (int k0 = 0; k0 < DMODEL; k0 += 32) {
    v16bf a = afrag_bf16(A, DMODEL, row0, k0);
#pragma unroll
    for (int t = 0; t < 4; ++t)
      acc[t] = wmma_bf16(a, bfrag_bf16(Wob, DMODEL, col0 + t * 16, k0), acc[t]);
  }

  int nlo   = lane & 15;
  int rbase = row0 + ((lane & 16) ? 8 : 0);
#pragma unroll
  for (int t = 0; t < 4; ++t) {
    int col = col0 + t * 16 + nlo;
    float bv = bo[col];
#pragma unroll
    for (int i = 0; i < 8; ++i)
      Y[(long)(rbase + i) * DMODEL + col] = acc[t][i] + bv;
  }
}

// ---------------------------------------------------------------------------
extern "C" void kernel_launch(void* const* d_in, const int* in_sizes, int n_in,
                              void* d_out, int out_size, void* d_ws, size_t ws_size,
                              hipStream_t stream) {
  const float* query = (const float*)d_in[0];
  const float* key   = (const float*)d_in[1];
  const float* value = (const float*)d_in[2];
  const float* cos_q = (const float*)d_in[3];
  const float* sin_q = (const float*)d_in[4];
  const float* cos_k = (const float*)d_in[5];
  const float* sin_k = (const float*)d_in[6];
  const unsigned char* mask = (const unsigned char*)d_in[7];
  const float* Wq = (const float*)d_in[8];
  const float* bq = (const float*)d_in[9];
  const float* Wk = (const float*)d_in[10];
  const float* bk = (const float*)d_in[11];
  const float* Wv = (const float*)d_in[12];
  const float* bv = (const float*)d_in[13];
  const float* Wo = (const float*)d_in[14];
  const float* bo = (const float*)d_in[15];

  // workspace carve: 4 x 16 MB bf16 head/activation buffers + 4 x 2 MB weights
  unsigned short* qh  = (unsigned short*)d_ws;
  unsigned short* kh  = qh + HBUF;
  unsigned short* vt  = kh + HBUF;
  unsigned short* oh  = vt + HBUF;
  unsigned short* wqb = oh + HBUF;
  unsigned short* wkb = wqb + WBUF;
  unsigned short* wvb = wkb + WBUF;
  unsigned short* wob = wvb + WBUF;

  dim3 blk(256);
  dim3 gconv((int)(WBUF / 8 / 256));     // 512 blocks
  dim3 gproj(ROWS / 128, DMODEL / 64);   // (64, 16)
  dim3 gattn(SEQ / 128, BATCH * NHEADS); // (16, 64)

  bf16conv_kernel<<<gconv, blk, 0, stream>>>(Wq, wqb, (int)(WBUF / 8));
  bf16conv_kernel<<<gconv, blk, 0, stream>>>(Wk, wkb, (int)(WBUF / 8));
  bf16conv_kernel<<<gconv, blk, 0, stream>>>(Wv, wvb, (int)(WBUF / 8));
  bf16conv_kernel<<<gconv, blk, 0, stream>>>(Wo, wob, (int)(WBUF / 8));

  proj_kernel<0><<<gproj, blk, 0, stream>>>(query, wqb, bq, cos_q, sin_q, qh);
  proj_kernel<1><<<gproj, blk, 0, stream>>>(key,   wkb, bk, cos_k, sin_k, kh);
  proj_kernel<2><<<gproj, blk, 0, stream>>>(value, wvb, bv, nullptr, nullptr, vt);
  attn_kernel<<<gattn, blk, 0, stream>>>(qh, kh, vt, mask, oh);
  oproj_kernel<<<gproj, blk, 0, stream>>>(oh, wob, bo, (float*)d_out);
}
